// AdaptedAttention_72541997629730
// MI455X (gfx1250) — compile-verified, matrix operand order checked by reading, not run on hardware
//
#include <hip/hip_runtime.h>
#include <hip/hip_bf16.h>

typedef unsigned short u16;
typedef __attribute__((ext_vector_type(16))) __bf16 v16bf;
typedef __attribute__((ext_vector_type(8)))  float  v8f;
typedef __attribute__((ext_vector_type(4)))  unsigned int v4u;

#define DMODEL 2048
#define SEQ    2048
#define HEADS  16
#define HDIM   128
#define LPROMPT 10

// ---------- helpers ----------
__device__ __forceinline__ u16 f2bf(float f) {
    unsigned int u = __float_as_uint(f);
    unsigned int r = u + 0x7FFFu + ((u >> 16) & 1u);   // round-to-nearest-even
    return (u16)(r >> 16);
}
__device__ __forceinline__ float bf2f(u16 h) {
    return __uint_as_float(((unsigned int)h) << 16);
}
__device__ __forceinline__ v8f vzero8() {
    v8f z = {0.f,0.f,0.f,0.f,0.f,0.f,0.f,0.f};
    return z;
}
union FragBF { v16bf v; v4u q[2]; };

__device__ __forceinline__ v8f wmma_bf16(v16bf a, v16bf b, v8f c) {
    return __builtin_amdgcn_wmma_f32_16x16x32_bf16(false, a, false, b, (short)0, c, false, false);
}

// async copy 16B/lane from global to LDS (gfx1250, ASYNCcnt-tracked)
__device__ __forceinline__ void async_g2l_b128(const u16* gsrc, void* ldst) {
    unsigned int loff = (unsigned int)(unsigned long long)ldst;  // LDS offset = addr[31:0]
    asm volatile("global_load_async_to_lds_b128 %0, %1, off"
                 :: "v"(loff), "v"(gsrc) : "memory");
}

// ---------- f32 -> bf16 conversion ----------
__global__ void k_cvt_bf16(const float* __restrict__ in, u16* __restrict__ out, int n) {
    int i = blockIdx.x * 256 + threadIdx.x;
    if (i < n) out[i] = f2bf(in[i]);
}

// ---------- WMMA GEMM: C[M,N] = A[M,K] * B[N,K]^T  (bf16 in, f32 out) ----------
// block = (32,4): 4 waves; block computes 64(M) x 64(N).
// B tiles (shared by all 4 waves) staged via async-to-LDS, double-buffered.
__global__ void k_gemm_bt(const u16* __restrict__ A, const u16* __restrict__ B,
                          float* __restrict__ C, int M, int N, int K) {
    __shared__ u16 Bsh[2][64][32];                  // 8 KB double buffer
    const int lane = threadIdx.x;                   // 0..31
    const int wv   = threadIdx.y;                   // 0..3
    const int tid  = wv * 32 + lane;                // 0..127
    const int nn   = lane & 15;
    const int half = lane >> 4;
    const int mt   = blockIdx.y * 4 + wv;           // 16-row tile (per wave)
    const int nt0  = blockIdx.x * 4;                // 4 shared 16-col tiles

    v8f acc[4];
    #pragma unroll
    for (int t = 0; t < 4; ++t) acc[t] = vzero8();

    const u16* arow = A + (size_t)(mt * 16 + nn) * K;

    // stage issue: 64 rows x 64B = 4KB per k-step; 2 async b128 per thread
    auto issue = [&](int k0, int buf) {
        #pragma unroll
        for (int s = 0; s < 2; ++s) {
            int i   = tid + s * 128;                // 0..255
            int row = i >> 2;                       // 0..63
            int seg = i & 3;                        // 0..3 (16B segments)
            const u16* g = B + (size_t)(nt0 * 16 + row) * K + k0 + seg * 8;
            async_g2l_b128(g, &Bsh[buf][row][seg * 8]);
        }
    };

    issue(0, 0);
    int buf = 0;
    for (int k0 = 0; k0 < K; k0 += 32) {
        if (k0 + 32 < K) {
            issue(k0 + 32, buf ^ 1);
            asm volatile("s_wait_asynccnt 0x2" ::: "memory");   // stage k0 landed
        } else {
            asm volatile("s_wait_asynccnt 0x0" ::: "memory");
        }
        __syncthreads();                            // all waves' stage visible

        __builtin_prefetch(arow + k0 + 256, 0, 0);  // global_prefetch_b8 (A stream)
        FragBF fa;
        fa.q[0] = *(const v4u*)(arow + k0 + half * 8);
        fa.q[1] = *(const v4u*)(arow + k0 + half * 8 + 16);
        #pragma unroll
        for (int t = 0; t < 4; ++t) {
            FragBF fb;
            fb.q[0] = *(const v4u*)(&Bsh[buf][t * 16 + nn][half * 8]);
            fb.q[1] = *(const v4u*)(&Bsh[buf][t * 16 + nn][half * 8 + 16]);
            acc[t] = wmma_bf16(fa.v, fb.v, acc[t]);
        }
        __syncthreads();                            // done reading buf before overwrite
        buf ^= 1;
    }
    #pragma unroll
    for (int t = 0; t < 4; ++t) {
        #pragma unroll
        for (int r = 0; r < 8; ++r) {
            int row = mt * 16 + r + half * 8;
            int col = (nt0 + t) * 16 + nn;
            C[(size_t)row * N + col] = acc[t][r];
        }
    }
}

// ---------- RoPE + convert to bf16 ----------
__global__ void k_rope_cvt(const float* __restrict__ in, u16* __restrict__ out,
                           const int* __restrict__ pos) {
    int i   = blockIdx.x * 256 + threadIdx.x;      // over SEQ*DMODEL
    int row = i >> 11;
    int col = i & (DMODEL - 1);
    int j   = col & (HDIM - 1);
    int jj  = j & 63;
    // inv_freq = 10000^(-2*jj/128)
    float invf = __expf(-(float)(2 * jj) * (9.210340371976184f / 128.0f));
    float ang  = (float)pos[row] * invf;
    float c = __cosf(ang), s = __sinf(ang);
    float x = in[i];
    float partner = (j < 64) ? -in[i + 64] : in[i - 64];
    out[i] = f2bf(x * c + partner * s);
}

// ---------- transpose f32[2048][2048] -> bf16[2048][2048] (V -> Vt) ----------
__global__ void k_transpose_cvt(const float* __restrict__ in, u16* __restrict__ out) {
    __shared__ u16 tile[32][33];
    int tx = threadIdx.x, ty = threadIdx.y;
    int bx = blockIdx.x * 32, by = blockIdx.y * 32;
    #pragma unroll
    for (int i = 0; i < 32; i += 8)
        tile[ty + i][tx] = f2bf(in[(size_t)(by + ty + i) * DMODEL + bx + tx]);
    __syncthreads();
    #pragma unroll
    for (int i = 0; i < 32; i += 8)
        out[(size_t)(bx + ty + i) * DMODEL + by + tx] = tile[tx][ty + i];
}

// ---------- flash attention (causal), 1 wave per (head, 16-query tile) ----------
__global__ void k_flash_attn(const u16* __restrict__ qb, const u16* __restrict__ kb_,
                             const u16* __restrict__ vt, float* __restrict__ ctx) {
    __shared__ u16 Pl[16][32];
    const int lane = threadIdx.x;
    const int nn   = lane & 15;
    const int half = lane >> 4;
    const int qt   = blockIdx.x;      // 0..127
    const int h    = blockIdx.y;      // 0..15
    const float scale = 0.08838834764831845f;   // 1/sqrt(128)

    // Q fragments: 4 chunks of K=32 covering head_dim 128
    FragBF qa[4];
    const u16* qrow = qb + (size_t)(qt * 16 + nn) * DMODEL + h * HDIM;
    #pragma unroll
    for (int kk = 0; kk < 4; ++kk) {
        qa[kk].q[0] = *(const v4u*)(qrow + kk * 32 + half * 8);
        qa[kk].q[1] = *(const v4u*)(qrow + kk * 32 + half * 8 + 16);
    }

    v8f acc[8];
    #pragma unroll
    for (int t = 0; t < 8; ++t) acc[t] = vzero8();
    float mrow[8], lrow[8];
    #pragma unroll
    for (int r = 0; r < 8; ++r) { mrow[r] = -3.0e38f; lrow[r] = 0.f; }

    const int limit = (qt + 1) * 16;
    for (int kbase = 0; kbase < limit; kbase += 32) {
        // ---- S = Q K^T for two key tiles (kbase..+15, kbase+16..+31) ----
        v8f s0 = vzero8(), s1 = vzero8();
        #pragma unroll
        for (int kk = 0; kk < 4; ++kk) {
            const u16* kr0 = kb_ + (size_t)(kbase + nn) * DMODEL + h * HDIM + kk * 32;
            const u16* kr1 = kb_ + (size_t)(kbase + 16 + nn) * DMODEL + h * HDIM + kk * 32;
            FragBF f0, f1;
            f0.q[0] = *(const v4u*)(kr0 + half * 8);
            f0.q[1] = *(const v4u*)(kr0 + half * 8 + 16);
            f1.q[0] = *(const v4u*)(kr1 + half * 8);
            f1.q[1] = *(const v4u*)(kr1 + half * 8 + 16);
            s0 = wmma_bf16(qa[kk].v, f0.v, s0);
            s1 = wmma_bf16(qa[kk].v, f1.v, s1);
        }
        // ---- online softmax over 32 keys ----
        float p0a[8], p1a[8], alpha[8];
        #pragma unroll
        for (int r = 0; r < 8; ++r) {
            int gm = qt * 16 + r + half * 8;
            float x0 = s0[r] * scale;
            float x1 = s1[r] * scale;
            if (kbase + nn > gm)      x0 = -3.0e38f;
            if (kbase + 16 + nn > gm) x1 = -3.0e38f;
            float rm = fmaxf(x0, x1);
            rm = fmaxf(rm, __shfl_xor(rm, 1, 32));
            rm = fmaxf(rm, __shfl_xor(rm, 2, 32));
            rm = fmaxf(rm, __shfl_xor(rm, 4, 32));
            rm = fmaxf(rm, __shfl_xor(rm, 8, 32));
            float nm = fmaxf(mrow[r], rm);
            float e0 = __expf(x0 - nm);
            float e1 = __expf(x1 - nm);
            float rs = e0 + e1;
            rs += __shfl_xor(rs, 1, 32);
            rs += __shfl_xor(rs, 2, 32);
            rs += __shfl_xor(rs, 4, 32);
            rs += __shfl_xor(rs, 8, 32);
            alpha[r] = __expf(mrow[r] - nm);
            lrow[r]  = lrow[r] * alpha[r] + rs;
            mrow[r]  = nm;
            p0a[r] = e0; p1a[r] = e1;
        }
        #pragma unroll
        for (int t = 0; t < 8; ++t)
            #pragma unroll
            for (int r = 0; r < 8; ++r) acc[t][r] *= alpha[r];

        // ---- stage P into LDS (C-layout -> A-layout transpose) ----
        #pragma unroll
        for (int r = 0; r < 8; ++r) {
            Pl[r + half * 8][nn]      = f2bf(p0a[r]);
            Pl[r + half * 8][16 + nn] = f2bf(p1a[r]);
        }
        __syncthreads();

        // ---- PV: acc += P(16x32) * Vt(32x128) ----
        FragBF pf;
        pf.q[0] = *(const v4u*)(&Pl[nn][half * 8]);
        pf.q[1] = *(const v4u*)(&Pl[nn][half * 8 + 16]);
        #pragma unroll
        for (int t = 0; t < 8; ++t) {
            const u16* vrow = vt + (size_t)(h * HDIM + t * 16 + nn) * SEQ + kbase;
            FragBF vf;
            vf.q[0] = *(const v4u*)(vrow + half * 8);
            vf.q[1] = *(const v4u*)(vrow + half * 8 + 16);
            acc[t] = wmma_bf16(pf.v, vf.v, acc[t]);
        }
        __syncthreads();
    }

    // ---- epilogue: normalize and store context (f32) ----
    #pragma unroll
    for (int t = 0; t < 8; ++t) {
        #pragma unroll
        for (int r = 0; r < 8; ++r) {
            float o = acc[t][r] / lrow[r];
            ctx[(size_t)(qt * 16 + r + half * 8) * DMODEL + h * HDIM + t * 16 + nn] = o;
        }
    }
}

// ---------- adapter K/V projections (tiny: 10 x 2048 x 2048, fp32) ----------
__global__ void k_adapter_kv(const float* __restrict__ prompt,
                             const float* __restrict__ Wk, const float* __restrict__ Wv,
                             float* __restrict__ ak, float* __restrict__ av) {
    int n = blockIdx.x * 128 + threadIdx.x;  // 0..2047
    int l = blockIdx.y;                      // 0..9
    const float* p  = prompt + (size_t)l * DMODEL;
    const float* wk = Wk + (size_t)n * DMODEL;
    const float* wv = Wv + (size_t)n * DMODEL;
    float sk = 0.f, sv = 0.f;
    for (int k = 0; k < DMODEL; ++k) {
        float pv = p[k];
        sk += pv * wk[k];
        sv += pv * wv[k];
    }
    ak[(size_t)l * DMODEL + n] = sk;
    av[(size_t)l * DMODEL + n] = sv;
}

// ---------- adapter attention: ctx += gate * softmax(q . ak / 4) @ av ----------
__global__ void k_adapter_apply(const u16* __restrict__ qrot,
                                const float* __restrict__ ak, const float* __restrict__ av,
                                const float* __restrict__ gatep, float* __restrict__ ctx) {
    __shared__ float wpart[4][LPROMPT];
    const int t   = threadIdx.x;       // 0..127 (head dim)
    const int row = blockIdx.x;        // 0..2047
    const int h   = blockIdx.y;        // 0..15
    const int base = h * HDIM + t;

    float qv = bf2f(qrot[(size_t)row * DMODEL + base]);
    #pragma unroll
    for (int l = 0; l < LPROMPT; ++l) {
        float prod = qv * ak[(size_t)l * DMODEL + base];
        prod += __shfl_xor(prod, 16, 32);
        prod += __shfl_xor(prod, 8, 32);
        prod += __shfl_xor(prod, 4, 32);
        prod += __shfl_xor(prod, 2, 32);
        prod += __shfl_xor(prod, 1, 32);
        if ((t & 31) == 0) wpart[t >> 5][l] = prod;
    }
    __syncthreads();
    float sc[LPROMPT];
    float mx = -3.0e38f;
    #pragma unroll
    for (int l = 0; l < LPROMPT; ++l) {
        float s = (wpart[0][l] + wpart[1][l] + wpart[2][l] + wpart[3][l]) * 0.25f; // 1/sqrt(16)
        sc[l] = s;
        mx = fmaxf(mx, s);
    }
    float den = 0.f;
    #pragma unroll
    for (int l = 0; l < LPROMPT; ++l) {
        float e = __expf(sc[l] - mx);
        sc[l] = e;
        den += e;
    }
    float accv = 0.f;
    #pragma unroll
    for (int l = 0; l < LPROMPT; ++l)
        accv += sc[l] * av[(size_t)l * DMODEL + base];
    ctx[(size_t)row * DMODEL + base] += gatep[0] * accv / den;
}

// ================== launcher ==================
extern "C" void kernel_launch(void* const* d_in, const int* in_sizes, int n_in,
                              void* d_out, int out_size, void* d_ws, size_t ws_size,
                              hipStream_t stream) {
    const float* hs     = (const float*)d_in[0];
    const float* Wq     = (const float*)d_in[1];
    const float* Wk     = (const float*)d_in[2];
    const float* Wv     = (const float*)d_in[3];
    const float* Wo     = (const float*)d_in[4];
    const float* prompt = (const float*)d_in[5];
    const float* gate   = (const float*)d_in[6];
    const int*   pos    = (const int*)d_in[7];
    float* out = (float*)d_out;

    char* ws = (char*)d_ws;
    const size_t MB = 1024ull * 1024ull;
    u16*   hsb  = (u16*)(ws + 0 * MB);      // 8 MB
    u16*   wqb  = (u16*)(ws + 8 * MB);      // 8 MB
    u16*   wkb  = (u16*)(ws + 16 * MB);     // 8 MB
    u16*   wvb  = (u16*)(ws + 24 * MB);     // 8 MB
    u16*   wob  = (u16*)(ws + 32 * MB);     // 8 MB
    float* qf   = (float*)(ws + 40 * MB);   // 16 MB
    float* kf   = (float*)(ws + 56 * MB);   // 16 MB
    float* vf   = (float*)(ws + 72 * MB);   // 16 MB
    u16*   qbf  = (u16*)(ws + 88 * MB);     // 8 MB
    u16*   kbf  = (u16*)(ws + 96 * MB);     // 8 MB
    u16*   vtb  = (u16*)(ws + 104 * MB);    // 8 MB
    float* ctxf = (float*)(ws + 112 * MB);  // 16 MB
    u16*   ctxb = (u16*)(ws + 128 * MB);    // 8 MB
    float* ak   = (float*)(ws + 136 * MB);  // 80 KB
    float* av   = (float*)(ws + 136 * MB + 128 * 1024);

    const int NELT = SEQ * DMODEL;          // 4M

    // 1) f32 -> bf16 conversions
    k_cvt_bf16<<<NELT / 256, 256, 0, stream>>>(hs, hsb, NELT);
    k_cvt_bf16<<<NELT / 256, 256, 0, stream>>>(Wq, wqb, NELT);
    k_cvt_bf16<<<NELT / 256, 256, 0, stream>>>(Wk, wkb, NELT);
    k_cvt_bf16<<<NELT / 256, 256, 0, stream>>>(Wv, wvb, NELT);
    k_cvt_bf16<<<NELT / 256, 256, 0, stream>>>(Wo, wob, NELT);

    // 2) Q/K/V projections (WMMA bf16 GEMMs, async-LDS staged B)
    dim3 gblk(32, 4);
    dim3 ggrd(DMODEL / 64, SEQ / 64);
    k_gemm_bt<<<ggrd, gblk, 0, stream>>>(hsb, wqb, qf, SEQ, DMODEL, DMODEL);
    k_gemm_bt<<<ggrd, gblk, 0, stream>>>(hsb, wkb, kf, SEQ, DMODEL, DMODEL);
    k_gemm_bt<<<ggrd, gblk, 0, stream>>>(hsb, wvb, vf, SEQ, DMODEL, DMODEL);

    // 3) RoPE + cvt for Q,K ; transpose-cvt for V
    k_rope_cvt<<<NELT / 256, 256, 0, stream>>>(qf, qbf, pos);
    k_rope_cvt<<<NELT / 256, 256, 0, stream>>>(kf, kbf, pos);
    k_transpose_cvt<<<dim3(DMODEL / 32, SEQ / 32), dim3(32, 8), 0, stream>>>(vf, vtb);

    // 4) causal flash attention (WMMA)
    k_flash_attn<<<dim3(SEQ / 16, HEADS), 32, 0, stream>>>(qbf, kbf, vtb, ctxf);

    // 5) adapter attention fused into ctx
    k_adapter_kv<<<dim3(DMODEL / 128, LPROMPT), 128, 0, stream>>>(prompt, Wk, Wv, ak, av);
    k_adapter_apply<<<dim3(SEQ, HEADS), 128, 0, stream>>>(qbf, ak, av, gate, ctxf);

    // 6) output projection: (ctx + a_ctx) @ Wo^T  (WMMA)
    k_cvt_bf16<<<NELT / 256, 256, 0, stream>>>(ctxf, ctxb, NELT);
    k_gemm_bt<<<ggrd, gblk, 0, stream>>>(ctxb, wob, out, SEQ, DMODEL, DMODEL);
}
